// AttentionBlockConv_6201932775591
// MI455X (gfx1250) — matrix operand order repaired and checked by reading, not compiled, not run on hardware
//
#include <hip/hip_runtime.h>

// ---------------------------------------------------------------------------
// Types for CDNA5 WMMA (wave32): D(16x16 f32) = A(16x32 bf16) x B(32x16 bf16) + C
// ---------------------------------------------------------------------------
typedef __attribute__((ext_vector_type(16))) __bf16 v16bf;
typedef __attribute__((ext_vector_type(8)))  __bf16 v8bf;
typedef __attribute__((ext_vector_type(8)))  float  v8f;

// A-matrix (16x32, 16-bit) per-lane layout (ISA 7.12.2):
// lane L holds row M = L&15; base = (L<16)?0:8
//   halves 0..7  -> K = base + h
//   halves 8..15 -> K = base + 16 + (h-8)
__device__ __forceinline__ v16bf load_a_frag(const __bf16* row, int base) {
    v8bf lo = *(const v8bf*)(row + base);
    v8bf hi = *(const v8bf*)(row + base + 16);
    v16bf r;
#pragma unroll
    for (int i = 0; i < 8; ++i) { r[i] = lo[i]; r[i + 8] = hi[i]; }
    return r;
}

// B-matrix (32x16, 16-bit) per-lane layout:
// lane L holds col N = L&15; off = (L<16)?0:8; halves 0..7 -> K = off + h, 8..15 -> K = off + 16 + (h-8)
__device__ __forceinline__ v16bf load_b_frag(const __bf16* p) {
    v8bf lo = *(const v8bf*)(p);
    v8bf hi = *(const v8bf*)(p + 16);
    v16bf r;
#pragma unroll
    for (int i = 0; i < 8; ++i) { r[i] = lo[i]; r[i + 8] = hi[i]; }
    return r;
}

__device__ __forceinline__ v8f wmma_bf16(v16bf a, v16bf b, v8f c) {
    return __builtin_amdgcn_wmma_f32_16x16x32_bf16(false, a, false, b, (short)0, c,
                                                   false, false);
}

// ---------------------------------------------------------------------------
// Problem constants
// ---------------------------------------------------------------------------
#define SIDE   64
#define NTOK   1024       // 32x32 tokens
#define CIN    256
#define CTOT   288        // 256 + 2*16 pos channels
#define RDIM   1152       // 288 * 4 taps
#define KDIM   256
#define OC3    768        // q|k|v stacked

// ---------------------------------------------------------------------------
// Kernel 1: convert weights to bf16
//   Wqkv[768][1152] : r = c*4 + ky*2 + kx  (natural flat order of (288,2,2))
//   Wup [4][256 out][256 in] : up_w layout is (in, out, ky, kx)
// ---------------------------------------------------------------------------
__global__ void k_prep(const float* __restrict__ qw, const float* __restrict__ kw,
                       const float* __restrict__ vw, const float* __restrict__ upw,
                       __bf16* __restrict__ wqkv, __bf16* __restrict__ wup) {
    int i = blockIdx.x * 256 + threadIdx.x;
    const int T1 = OC3 * RDIM;
    if (i < T1) {
        int o = i / RDIM, r = i % RDIM;
        float v;
        if (o < 256)      v = qw[o * RDIM + r];
        else if (o < 512) v = kw[(o - 256) * RDIM + r];
        else              v = vw[(o - 512) * RDIM + r];
        wqkv[i] = (__bf16)v;
    } else {
        int j = i - T1;                       // [tap][o][in]
        if (j < 4 * 256 * 256) {
            int in = j & 255, o = (j >> 8) & 255, t = j >> 16;
            int ky = t >> 1, kx = t & 1;
            wup[j] = (__bf16)upw[((in * 256 + o) * 2 + ky) * 2 + kx];
        }
    }
}

// ---------------------------------------------------------------------------
// Kernel 2: im2col (token-major patches), pos-enc channels computed inline.
//   Pt[b][n][r], r = c*4 + ky*2 + kx ; input pixel (2*oy-1+ky, 2*ox-1+kx)
// ---------------------------------------------------------------------------
__device__ __forceinline__ float pos_val(int p, int e) {
    if (p == 0) return 0.f;
    float expo  = (float)(2 * (e >> 1)) * (1.0f / 16.0f);
    float denom = __powf(10000.0f, expo);
    float ang   = (float)p / denom;
    return (e & 1) ? __cosf(ang) : __sinf(ang);
}

__global__ void k_im2col(const float* __restrict__ x, __bf16* __restrict__ pt) {
    int i = blockIdx.x * 256 + threadIdx.x;      // b*1024*288 + n*288 + c
    if (i >= 32 * NTOK * CTOT) return;
    int c = i % CTOT;
    int n = (i / CTOT) & (NTOK - 1);
    int b = i / (CTOT * NTOK);
    int oy = n >> 5, ox = n & 31;
    __bf16* dst = pt + ((size_t)(b * NTOK + n)) * RDIM + c * 4;
#pragma unroll
    for (int t = 0; t < 4; ++t) {
        int ky = t >> 1, kx = t & 1;
        int iy = 2 * oy - 1 + ky, ix = 2 * ox - 1 + kx;
        float v = 0.f;
        if (iy >= 0 && ix >= 0) {
            if (c < CIN) v = x[(((size_t)b * CIN + c) * SIDE + iy) * SIDE + ix];
            else {
                int e = c - CIN;
                v = (e < 16) ? pos_val(iy, e) : pos_val(ix, e - 16);
            }
        }
        dst[t] = (__bf16)v;
    }
}

// ---------------------------------------------------------------------------
// Kernel 3: QKV projection GEMM: C[768,1024] = Wqkv x Pt[b], bias + ReLU.
//   Q,K written token-major [b][n][256]; V written dim-major [b][256][n].
//   One wave -> 16 out-channels x 64 tokens (A fragment reused over 4 B tiles).
// ---------------------------------------------------------------------------
__global__ void k_qkv(const __bf16* __restrict__ wqkv, const __bf16* __restrict__ pt,
                      const float* __restrict__ qb, const float* __restrict__ kb,
                      const float* __restrict__ vb,
                      __bf16* __restrict__ qt, __bf16* __restrict__ kt,
                      __bf16* __restrict__ vm) {
    int lane = threadIdx.x & 31;
    int wg   = blockIdx.x * 8 + (threadIdx.x >> 5);   // 24576 waves
    int ng     = wg & 15;
    int octile = (wg >> 4) % 48;
    int b      = wg / (16 * 48);
    int oc0 = octile * 16, n0 = ng * 64;
    int lo = lane & 15, hi = lane >> 4;
    int abase = hi * 8, boff = hi * 8;

    v8f acc[4];
#pragma unroll
    for (int t = 0; t < 4; ++t)
#pragma unroll
        for (int j = 0; j < 8; ++j) acc[t][j] = 0.f;

    const __bf16* arow = wqkv + (size_t)(oc0 + lo) * RDIM;
    const __bf16* brow[4];
#pragma unroll
    for (int t = 0; t < 4; ++t)
        brow[t] = pt + ((size_t)(b * NTOK) + n0 + t * 16 + lo) * RDIM + boff;

    for (int kc = 0; kc < RDIM / 32; ++kc) {
        int r0 = kc * 32;
        v16bf a = load_a_frag(arow + r0, abase);
#pragma unroll
        for (int t = 0; t < 4; ++t) {
            v16bf bb = load_b_frag(brow[t] + r0);
            acc[t] = wmma_bf16(a, bb, acc[t]);
        }
    }

    int which = oc0 >> 8;          // 0=Q 1=K 2=V
    int ocl   = oc0 & 255;
    const float* bias = (which == 0) ? qb : (which == 1) ? kb : vb;
    if (which < 2) {
        __bf16* dst = (which == 0) ? qt : kt;
#pragma unroll
        for (int t = 0; t < 4; ++t) {
            int n = n0 + t * 16 + lo;
            __bf16* p = dst + ((size_t)(b * NTOK) + n) * KDIM + ocl + abase;
            v8bf pk;
#pragma unroll
            for (int j = 0; j < 8; ++j) {
                float v = acc[t][j] + bias[ocl + abase + j];
                pk[j] = (__bf16)fmaxf(v, 0.f);
            }
            *(v8bf*)p = pk;                    // 16B packed store
        }
    } else {
#pragma unroll
        for (int t = 0; t < 4; ++t) {
            int n = n0 + t * 16 + lo;
#pragma unroll
            for (int j = 0; j < 8; ++j) {
                int oc  = ocl + abase + j;
                float v = acc[t][j] + bias[oc];
                vm[((size_t)(b * 256) + oc) * NTOK + n] = (__bf16)fmaxf(v, 0.f);
            }
        }
    }
}

// ---------------------------------------------------------------------------
// Kernel 4: causal flash attention. One wave = 16 query tokens, full 256 vdim.
//   S^T = K^T Q  (D layout of S^T == A layout needed for P x V^T: no shuffles)
// ---------------------------------------------------------------------------
__global__ void k_attn(const __bf16* __restrict__ qt, const __bf16* __restrict__ kt,
                       const __bf16* __restrict__ vm, __bf16* __restrict__ o1) {
    int lane = threadIdx.x & 31;
    int wg   = blockIdx.x * 8 + (threadIdx.x >> 5);   // 2048 waves
    int b = wg >> 6, qtile = wg & 63;
    int n0 = qtile * 16;
    int lo = lane & 15, hi = lane >> 4;
    int abase = hi * 8, boff = hi * 8;
    int nq = n0 + lo;                                 // this lane's query index

    // Preload Q B-fragments (K-dim = 256 -> 8 chunks of 32)
    const __bf16* qrow = qt + ((size_t)(b * NTOK) + n0 + lo) * KDIM + boff;
    v16bf qf[8];
#pragma unroll
    for (int c = 0; c < 8; ++c) qf[c] = load_b_frag(qrow + c * 32);

    v8f oacc[16];
#pragma unroll
    for (int t = 0; t < 16; ++t)
#pragma unroll
        for (int j = 0; j < 8; ++j) oacc[t][j] = 0.f;

    float run_m = -1e30f, run_l = 0.f;
    int nblocks = ((n0 + 15) >> 5) + 1;

    for (int kb = 0; kb < nblocks; ++kb) {
        int m0 = kb * 32;
        v8f s0, s1;
#pragma unroll
        for (int j = 0; j < 8; ++j) { s0[j] = 0.f; s1[j] = 0.f; }

        const __bf16* kr0 = kt + ((size_t)(b * NTOK) + m0 + lo) * KDIM;
        const __bf16* kr1 = kr0 + 16 * KDIM;
#pragma unroll
        for (int c = 0; c < 8; ++c) {
            v16bf a0 = load_a_frag(kr0 + c * 32, abase);
            s0 = wmma_bf16(a0, qf[c], s0);
            v16bf a1 = load_a_frag(kr1 + c * 32, abase);
            s1 = wmma_bf16(a1, qf[c], s1);
        }

        // causal mask (keep m <= n), scale by 1/sqrt(256)
        float bm = -1e30f;
#pragma unroll
        for (int j = 0; j < 8; ++j) {
            int m_0 = m0 + abase + j;
            int m_1 = m_0 + 16;
            s0[j] = (m_0 <= nq) ? s0[j] * 0.0625f : -1e30f;
            s1[j] = (m_1 <= nq) ? s1[j] * 0.0625f : -1e30f;
            bm = fmaxf(bm, fmaxf(s0[j], s1[j]));
        }
        bm = fmaxf(bm, __shfl_xor(bm, 16, 32));       // row lives in lanes L, L^16

        float newm  = fmaxf(run_m, bm);
        float alpha = __expf(run_m - newm);
        float rs = 0.f;
        v16bf pa;                                     // == A fragment of P(16x32)
#pragma unroll
        for (int j = 0; j < 8; ++j) {
            float p0 = __expf(s0[j] - newm);
            float p1 = __expf(s1[j] - newm);
            rs += p0 + p1;
            pa[j]     = (__bf16)p0;
            pa[j + 8] = (__bf16)p1;
        }
        rs += __shfl_xor(rs, 16, 32);
        run_l = run_l * alpha + rs;
        run_m = newm;

        // broadcast per-row alpha to O-accumulator row distribution (n = abase+j)
        float av[8];
#pragma unroll
        for (int j = 0; j < 8; ++j) av[j] = __shfl(alpha, abase + j, 32);

        const __bf16* vrow = vm + ((size_t)(b * 256) + lo) * NTOK + m0 + boff;
#pragma unroll
        for (int t = 0; t < 16; ++t) {
#pragma unroll
            for (int j = 0; j < 8; ++j) oacc[t][j] *= av[j];
            v16bf vf = load_b_frag(vrow + (size_t)t * 16 * NTOK);
            oacc[t] = wmma_bf16(pa, vf, oacc[t]);
        }
    }

    float inv[8];
#pragma unroll
    for (int j = 0; j < 8; ++j) inv[j] = 1.0f / __shfl(run_l, abase + j, 32);

#pragma unroll
    for (int t = 0; t < 16; ++t) {
        int v = t * 16 + lo;
#pragma unroll
        for (int j = 0; j < 8; ++j) {
            int n = n0 + abase + j;
            o1[((size_t)(b * NTOK) + n) * 256 + v] = (__bf16)(oacc[t][j] * inv[j]);
        }
    }
}

// ---------------------------------------------------------------------------
// Kernel 5: ConvTranspose2d(k=2,s=2) as 4 tap-GEMMs + residual + bias.
//   out[b][o][2iy+ky][2ix+kx] = x + up_b[o] + sum_in Wup[t][o][in]*o1t[b][n][in]
// ---------------------------------------------------------------------------
__global__ void k_upconv(const __bf16* __restrict__ wup, const __bf16* __restrict__ o1,
                         const float* __restrict__ x, const float* __restrict__ upb,
                         float* __restrict__ out) {
    int lane = threadIdx.x & 31;
    int wg   = blockIdx.x * 8 + (threadIdx.x >> 5);   // 32768 waves
    int ng     = wg & 15;
    int octile = (wg >> 4) & 15;
    int tap    = (wg >> 8) & 3;
    int b      = wg >> 10;
    int oc0 = octile * 16, n0 = ng * 64;
    int lo = lane & 15, hi = lane >> 4;
    int abase = hi * 8, boff = hi * 8;

    v8f acc[4];
#pragma unroll
    for (int t = 0; t < 4; ++t)
#pragma unroll
        for (int j = 0; j < 8; ++j) acc[t][j] = 0.f;

    const __bf16* arow = wup + ((size_t)tap * 256 + oc0 + lo) * 256;
    for (int kc = 0; kc < 8; ++kc) {
        v16bf a = load_a_frag(arow + kc * 32, abase);
#pragma unroll
        for (int t = 0; t < 4; ++t) {
            const __bf16* bp =
                o1 + ((size_t)(b * NTOK) + n0 + t * 16 + lo) * 256 + kc * 32 + boff;
            v16bf bb = load_b_frag(bp);
            acc[t] = wmma_bf16(a, bb, acc[t]);
        }
    }

    int ky = tap >> 1, kx = tap & 1;
#pragma unroll
    for (int t = 0; t < 4; ++t) {
        int n  = n0 + t * 16 + lo;
        int iy = n >> 5, ix = n & 31;
        int oy = 2 * iy + ky, ox = 2 * ix + kx;
#pragma unroll
        for (int j = 0; j < 8; ++j) {
            int oc = oc0 + abase + j;
            size_t idx = (((size_t)b * 256 + oc) * SIDE + oy) * SIDE + ox;
            out[idx] = x[idx] + upb[oc] + acc[t][j];
        }
    }
}

// ---------------------------------------------------------------------------
// Launch
// ---------------------------------------------------------------------------
extern "C" void kernel_launch(void* const* d_in, const int* in_sizes, int n_in,
                              void* d_out, int out_size, void* d_ws, size_t ws_size,
                              hipStream_t stream) {
    const float* x   = (const float*)d_in[0];
    const float* qw  = (const float*)d_in[1];
    const float* qb  = (const float*)d_in[2];
    const float* kw  = (const float*)d_in[3];
    const float* kb  = (const float*)d_in[4];
    const float* vw  = (const float*)d_in[5];
    const float* vb  = (const float*)d_in[6];
    const float* upw = (const float*)d_in[7];
    const float* upb = (const float*)d_in[8];
    float* out = (float*)d_out;

    char* ws = (char*)d_ws;
    // byte offsets (256B aligned)
    __bf16* wqkv = (__bf16*)(ws + 0);            //  768*1152*2        = 1,769,472
    __bf16* wup  = (__bf16*)(ws + 1769472);      //  4*256*256*2       =   524,288
    __bf16* pt   = (__bf16*)(ws + 2293760);      //  32*1024*1152*2    = 75,497,472
    __bf16* qt   = (__bf16*)(ws + 77791232);     //  32*1024*256*2     = 16,777,216
    __bf16* kt   = (__bf16*)(ws + 94568448);
    __bf16* vm   = (__bf16*)(ws + 111345664);
    __bf16* o1   = (__bf16*)(ws + 128122880);    // end = 144,900,096 bytes

    k_prep<<<4480, 256, 0, stream>>>(qw, kw, vw, upw, wqkv, wup);
    k_im2col<<<36864, 256, 0, stream>>>(x, pt);
    k_qkv<<<3072, 256, 0, stream>>>(wqkv, pt, qb, kb, vb, qt, kt, vm);
    k_attn<<<256, 256, 0, stream>>>(qt, kt, vm, o1);
    k_upconv<<<4096, 256, 0, stream>>>(wup, o1, x, upb, out);
}
